// EMGMambaRegressor_52089363366466
// MI455X (gfx1250) — matrix operand order repaired and checked by reading, not compiled, not run on hardware
//
#include <hip/hip_runtime.h>
#include <math.h>

#define TB   1024
#define BATCH 4
#define CINC 12
#define HID  256
#define DIN  512
#define ROWS (BATCH*TB)   // 4096

typedef __attribute__((ext_vector_type(16))) _Float16 v16h;
typedef __attribute__((ext_vector_type(8)))  _Float16 h8;
typedef __attribute__((ext_vector_type(8)))  float    v8f;

__device__ __forceinline__ float sigmoidf_(float x){ return 1.0f/(1.0f+__expf(-x)); }
__device__ __forceinline__ float siluf_(float x){ return x*sigmoidf_(x); }
__device__ __forceinline__ float softplusf_(float x){ return (x>20.0f)? x : log1pf(__expf(x)); }

// ---------------------------------------------------------------------------
// f32 -> f16 elementwise (n multiple of 8)
// ---------------------------------------------------------------------------
__global__ void k_cvt16(const float* __restrict__ in, _Float16* __restrict__ outp, int n)
{
  int base = (blockIdx.x*blockDim.x + threadIdx.x)*8;
  if (base >= n) return;
  float4 a = *(const float4*)(in+base);
  float4 b = *(const float4*)(in+base+4);
  h8 o;
  o[0]=(_Float16)a.x; o[1]=(_Float16)a.y; o[2]=(_Float16)a.z; o[3]=(_Float16)a.w;
  o[4]=(_Float16)b.x; o[5]=(_Float16)b.y; o[6]=(_Float16)b.z; o[7]=(_Float16)b.w;
  *(h8*)(outp+base) = o;
}

// ---------------------------------------------------------------------------
// Front-end: sum of k=3/5/7 convs over CIN=12, /3.  x:[B,T,12] -> h:[B,T,256]
// ---------------------------------------------------------------------------
__global__ void k_front(const float* __restrict__ x,
                        const float* __restrict__ w3, const float* __restrict__ b3,
                        const float* __restrict__ w5, const float* __restrict__ b5,
                        const float* __restrict__ w7, const float* __restrict__ b7,
                        float* __restrict__ h)
{
  int idx = blockIdx.x*blockDim.x + threadIdx.x;
  if (idx >= ROWS*HID) return;
  int o = idx % HID;
  int t = (idx / HID) % TB;
  int b = idx / (HID*TB);
  const float* xb = x + (size_t)b*TB*CINC;
  float acc = b3[o] + b5[o] + b7[o];
  #pragma unroll
  for (int j=0;j<3;j++){ int tt=t-1+j; if(tt>=0&&tt<TB){ const float* xr=xb+(size_t)tt*CINC; const float* wr=w3+(size_t)o*CINC*3;
      for(int ci=0;ci<CINC;ci++) acc += xr[ci]*wr[ci*3+j]; } }
  #pragma unroll
  for (int j=0;j<5;j++){ int tt=t-2+j; if(tt>=0&&tt<TB){ const float* xr=xb+(size_t)tt*CINC; const float* wr=w5+(size_t)o*CINC*5;
      for(int ci=0;ci<CINC;ci++) acc += xr[ci]*wr[ci*5+j]; } }
  #pragma unroll
  for (int j=0;j<7;j++){ int tt=t-3+j; if(tt>=0&&tt<TB){ const float* xr=xb+(size_t)tt*CINC; const float* wr=w7+(size_t)o*CINC*7;
      for(int ci=0;ci<CINC;ci++) acc += xr[ci]*wr[ci*7+j]; } }
  h[idx] = acc * (1.0f/3.0f);
}

// ---------------------------------------------------------------------------
// Async double-buffered WMMA GEMM on f16 operands.
// Cout[M,N] = A16[M,K] * W16[N,K]^T (+ Cin).  Requires M%64==0, N%128==0, K%32==0.
// Block: 256 thr (8 waves), tile 64x128, BK=32. Wave: 16 rows x 64 cols (4 wmma).
// Staging via GLOBAL_LOAD_ASYNC_TO_LDS_B128 (ASYNCcnt), ping-pong LDS.
// ---------------------------------------------------------------------------
__global__ __launch_bounds__(256)
void k_gemm_async(const _Float16* __restrict__ A, const _Float16* __restrict__ W,
                  const float* __restrict__ Cin, float* __restrict__ Cout,
                  int M, int N, int K)
{
  __shared__ _Float16 sA[2][64*32];
  __shared__ _Float16 sW[2][128*32];
  const int tid  = threadIdx.x;
  const int mBase = blockIdx.x * 64;
  const int nBase = blockIdx.y * 128;
  const int wid  = tid >> 5, lane = tid & 31;
  const int half = lane >> 4, nl = lane & 15;
  const int wr = wid & 3;     // wave row (16 rows)
  const int wc = wid >> 2;    // wave col (64 cols)

  const int r  = tid >> 2;        // 0..63
  const int kg = (tid & 3) * 8;   // half-elements: 0,8,16,24 (16B chunks)

  // global source addresses (bytes advance by 64 per K-step of 32 halves)
  const unsigned long long gA  = (unsigned long long)(A + (size_t)(mBase + r)*K + kg);
  const unsigned long long gW0 = (unsigned long long)(W + (size_t)(nBase + r)*K + kg);
  const unsigned long long gW1 = (unsigned long long)(W + (size_t)(nBase + 64 + r)*K + kg);
  unsigned ldsA0  = (unsigned)(unsigned long long)&sA[0][r*32 + kg];
  unsigned ldsA1  = (unsigned)(unsigned long long)&sA[1][r*32 + kg];
  unsigned ldsW00 = (unsigned)(unsigned long long)&sW[0][r*32 + kg];
  unsigned ldsW01 = (unsigned)(unsigned long long)&sW[1][r*32 + kg];
  unsigned ldsW10 = (unsigned)(unsigned long long)&sW[0][(r+64)*32 + kg];
  unsigned ldsW11 = (unsigned)(unsigned long long)&sW[1][(r+64)*32 + kg];

  auto stage = [&](unsigned long long koffB, int buf){
    unsigned la  = buf ? ldsA1  : ldsA0;
    unsigned lw0 = buf ? ldsW01 : ldsW00;
    unsigned lw1 = buf ? ldsW11 : ldsW10;
    unsigned long long a0 = gA  + koffB;
    unsigned long long a1 = gW0 + koffB;
    unsigned long long a2 = gW1 + koffB;
    asm volatile("global_load_async_to_lds_b128 %0, %1, off" :: "v"(la),  "v"(a0) : "memory");
    asm volatile("global_load_async_to_lds_b128 %0, %1, off" :: "v"(lw0), "v"(a1) : "memory");
    asm volatile("global_load_async_to_lds_b128 %0, %1, off" :: "v"(lw1), "v"(a2) : "memory");
  };

  v8f acc[4] = {{},{},{},{}};
  if (Cin){
    #pragma unroll
    for (int tl=0; tl<4; tl++){
      #pragma unroll
      for (int v=0;v<8;v++){
        int gm = mBase + wr*16 + half*8 + v;
        int gn = nBase + wc*64 + tl*16 + nl;
        acc[tl][v] = Cin[(size_t)gm*N + gn];
      }
    }
  }

  const int nk = K/32;
  stage(0, 0);
  for (int s=0; s<nk; s++){
    int buf = s & 1;
    if (s+1 < nk){
      stage((unsigned long long)(s+1)*64, (s+1)&1);
      asm volatile("s_wait_asynccnt 0x3" ::: "memory");   // our 3 loads for stage s are done
    } else {
      asm volatile("s_wait_asynccnt 0x0" ::: "memory");
    }
    __syncthreads();

    v16h af;
    {
      const h8 a0 = *(const h8*)&sA[buf][(wr*16 + nl)*32 + half*16];
      const h8 a1 = *(const h8*)&sA[buf][(wr*16 + nl)*32 + half*16 + 8];
      #pragma unroll
      for (int i=0;i<8;i++){ af[i]=a0[i]; af[i+8]=a1[i]; }
    }
    #pragma unroll
    for (int tl=0; tl<4; tl++){
      int bn = wc*64 + tl*16 + nl;
      const h8 b0 = *(const h8*)&sW[buf][bn*32 + half*16];
      const h8 b1 = *(const h8*)&sW[buf][bn*32 + half*16 + 8];
      v16h bf;
      #pragma unroll
      for (int i=0;i<8;i++){ bf[i]=b0[i]; bf[i+8]=b1[i]; }
      acc[tl] = __builtin_amdgcn_wmma_f32_16x16x32_f16(false, af, false, bf, (short)0, acc[tl], false, false);
    }
    __syncthreads();
  }

  #pragma unroll
  for (int tl=0; tl<4; tl++){
    #pragma unroll
    for (int v=0;v<8;v++){
      int gm = mBase + wr*16 + half*8 + v;
      int gn = nBase + wc*64 + tl*16 + nl;
      Cout[(size_t)gm*N + gn] = acc[tl][v];
    }
  }
}

// ---------------------------------------------------------------------------
// Synchronous WMMA GEMM with guards (for ragged N=48 case). f32 in, f16 WMMA.
// Block 256 thr, 64x64 tile, BK=32; wave: 16x32 (2 wmma).
// ---------------------------------------------------------------------------
__global__ __launch_bounds__(256)
void k_gemm_sync(const float* __restrict__ A, const float* __restrict__ W,
                 float* __restrict__ Cout, int M, int N, int K)
{
  __shared__ _Float16 sA[64*32];
  __shared__ _Float16 sW[64*32];
  const int tid  = threadIdx.x;
  const int mBase = blockIdx.x * 64;
  const int nBase = blockIdx.y * 64;
  const int wid  = tid >> 5, lane = tid & 31;
  const int half = lane >> 4, nl = lane & 15;
  const int wr = wid & 3, wc = wid >> 2;

  v8f acc0 = {}, acc1 = {};
  const int r  = tid >> 2;
  const int kg = (tid & 3) * 8;

  for (int kk=0; kk<K; kk+=32){
    {
      const float* ap = A + (size_t)(mBase + r)*K + kk + kg;
      float4 a0 = *(const float4*)ap;
      float4 a1 = *(const float4*)(ap+4);
      _Float16* dst = &sA[r*32 + kg];
      dst[0]=(_Float16)a0.x; dst[1]=(_Float16)a0.y; dst[2]=(_Float16)a0.z; dst[3]=(_Float16)a0.w;
      dst[4]=(_Float16)a1.x; dst[5]=(_Float16)a1.y; dst[6]=(_Float16)a1.z; dst[7]=(_Float16)a1.w;
    }
    {
      int n = nBase + r;
      _Float16* dst = &sW[r*32 + kg];
      if (n < N){
        const float* wp = W + (size_t)n*K + kk + kg;
        float4 w0 = *(const float4*)wp;
        float4 w1 = *(const float4*)(wp+4);
        dst[0]=(_Float16)w0.x; dst[1]=(_Float16)w0.y; dst[2]=(_Float16)w0.z; dst[3]=(_Float16)w0.w;
        dst[4]=(_Float16)w1.x; dst[5]=(_Float16)w1.y; dst[6]=(_Float16)w1.z; dst[7]=(_Float16)w1.w;
      } else {
        #pragma unroll
        for (int i=0;i<8;i++) dst[i] = (_Float16)0.0f;
      }
    }
    __syncthreads();
    v16h af, bf0, bf1;
    {
      const h8 a0 = *(const h8*)&sA[(wr*16 + nl)*32 + half*16];
      const h8 a1 = *(const h8*)&sA[(wr*16 + nl)*32 + half*16 + 8];
      #pragma unroll
      for (int i=0;i<8;i++){ af[i]=a0[i]; af[i+8]=a1[i]; }
      int bn = wc*32 + nl;
      const h8 b00 = *(const h8*)&sW[bn*32 + half*16];
      const h8 b01 = *(const h8*)&sW[bn*32 + half*16 + 8];
      #pragma unroll
      for (int i=0;i<8;i++){ bf0[i]=b00[i]; bf0[i+8]=b01[i]; }
      const h8 b10 = *(const h8*)&sW[(bn+16)*32 + half*16];
      const h8 b11 = *(const h8*)&sW[(bn+16)*32 + half*16 + 8];
      #pragma unroll
      for (int i=0;i<8;i++){ bf1[i]=b10[i]; bf1[i+8]=b11[i]; }
    }
    acc0 = __builtin_amdgcn_wmma_f32_16x16x32_f16(false, af, false, bf0, (short)0, acc0, false, false);
    acc1 = __builtin_amdgcn_wmma_f32_16x16x32_f16(false, af, false, bf1, (short)0, acc1, false, false);
    __syncthreads();
  }

  #pragma unroll
  for (int v=0;v<8;v++){
    int gm = mBase + wr*16 + half*8 + v;
    int gn0 = nBase + wc*32 + nl;
    int gn1 = gn0 + 16;
    if (gn0 < N) Cout[(size_t)gm*N + gn0] = acc0[v];
    if (gn1 < N) Cout[(size_t)gm*N + gn1] = acc1[v];
  }
}

// ---------------------------------------------------------------------------
// Depthwise causal conv (K=4) over time + SiLU. xi = xz[:, :512]
// ---------------------------------------------------------------------------
__global__ void k_mconv(const float* __restrict__ xz, const float* __restrict__ cw,
                        const float* __restrict__ cb, float* __restrict__ xc)
{
  int idx = blockIdx.x*blockDim.x + threadIdx.x;
  if (idx >= ROWS*DIN) return;
  int d = idx % DIN;
  int row = idx / DIN;
  int t = row % TB;
  float acc = cb[d];
  #pragma unroll
  for (int j=0;j<4;j++){
    int tt = t-3+j;
    if (tt >= 0) acc += xz[(size_t)(row + (tt - t))*(2*DIN) + d] * cw[d*4+j];
  }
  xc[idx] = siluf_(acc);
}

// ---------------------------------------------------------------------------
// dt = softplus(xdbl[:, :16] @ dt_w^T + dt_b)
// ---------------------------------------------------------------------------
__global__ void k_dt(const float* __restrict__ xdbl, const float* __restrict__ dtw,
                     const float* __restrict__ dtb, float* __restrict__ dt)
{
  int idx = blockIdx.x*blockDim.x + threadIdx.x;
  if (idx >= ROWS*DIN) return;
  int d = idx % DIN;
  int row = idx / DIN;
  const float* xr = xdbl + (size_t)row*48;
  const float* wr = dtw + d*16;
  float acc = dtb[d];
  #pragma unroll
  for (int rr=0;rr<16;rr++) acc += xr[rr]*wr[rr];
  dt[idx] = softplusf_(acc);
}

// ---------------------------------------------------------------------------
// Selective scan: lane = state n (16 lanes/channel, 2 channels/wave).
// yz = (scan + xc*Dp) * silu(z)
// ---------------------------------------------------------------------------
__global__ __launch_bounds__(256)
void k_scan(const float* __restrict__ xz, const float* __restrict__ xc,
            const float* __restrict__ dt, const float* __restrict__ xdbl,
            const float* __restrict__ A_log, const float* __restrict__ Dp,
            float* __restrict__ yz)
{
  int tid = threadIdx.x;
  int lane = tid & 31;
  int half = lane >> 4;
  int n = lane & 15;
  int ch = blockIdx.x*16 + (tid>>5)*2 + half;   // 0..2047
  int b = ch >> 9;
  int d = ch & 511;
  float Ad  = -__expf(A_log[d*16 + n]);
  float Dpd = Dp[d];
  float hst = 0.0f;
  size_t rowBase = (size_t)b * TB;
  for (int t=0;t<TB;t++){
    size_t row = rowBase + t;
    if ((t & 7) == 0 && t + 8 < TB){
      __builtin_prefetch(&dt[(row+8)*DIN + d], 0, 0);
      __builtin_prefetch(&xdbl[(row+8)*48 + n], 0, 0);
    }
    float dtv = dt[row*DIN + d];
    float xcv = xc[row*DIN + d];
    float Bv  = xdbl[row*48 + 16 + n];
    float Cv  = xdbl[row*48 + 32 + n];
    float dA  = __expf(dtv * Ad);
    hst = dA*hst + dtv*Bv*xcv;
    float contrib = hst * Cv;
    contrib += __shfl_xor(contrib, 8, 16);
    contrib += __shfl_xor(contrib, 4, 16);
    contrib += __shfl_xor(contrib, 2, 16);
    contrib += __shfl_xor(contrib, 1, 16);
    if (n == 0){
      float y = contrib + xcv * Dpd;
      float z = xz[row*(2*DIN) + DIN + d];
      yz[row*DIN + d] = y * siluf_(z);
    }
  }
}

// ---------------------------------------------------------------------------
// LayerNorm over HID=256 per row
// ---------------------------------------------------------------------------
__global__ void k_ln(const float* __restrict__ tin, const float* __restrict__ g,
                     const float* __restrict__ bb, float* __restrict__ tout)
{
  __shared__ float red[256];
  int row = blockIdx.x; int c = threadIdx.x;
  float v = tin[(size_t)row*HID + c];
  red[c] = v; __syncthreads();
  for (int s=128;s>0;s>>=1){ if (c<s) red[c]+=red[c+s]; __syncthreads(); }
  float mean = red[0] * (1.0f/HID);
  __syncthreads();
  float dv = v - mean;
  red[c] = dv*dv; __syncthreads();
  for (int s=128;s>0;s>>=1){ if (c<s) red[c]+=red[c+s]; __syncthreads(); }
  float var = red[0] * (1.0f/HID);
  tout[(size_t)row*HID + c] = dv * rsqrtf(var + 1e-5f) * g[c] + bb[c];
}

// ---------------------------------------------------------------------------
// CBAM kernels
// ---------------------------------------------------------------------------
__global__ void k_chstat(const float* __restrict__ ln,
                         float* __restrict__ mc, float* __restrict__ mx)
{
  __shared__ float ssum[256]; __shared__ float smax[256];
  int b = blockIdx.x >> 8;
  int c = blockIdx.x & 255;
  int tid = threadIdx.x;
  float s = 0.0f, m = -1e30f;
  #pragma unroll
  for (int i=0;i<4;i++){
    int t = tid + i*256;
    float v = ln[((size_t)(b*TB + t))*HID + c];
    s += v; m = fmaxf(m, v);
  }
  ssum[tid]=s; smax[tid]=m; __syncthreads();
  for (int st=128; st>0; st>>=1){
    if (tid<st){ ssum[tid]+=ssum[tid+st]; smax[tid]=fmaxf(smax[tid],smax[tid+st]); }
    __syncthreads();
  }
  if (tid==0){ mc[b*HID+c]=ssum[0]*(1.0f/TB); mx[b*HID+c]=smax[0]; }
}

__global__ void k_camlp(const float* __restrict__ mc, const float* __restrict__ mx,
                        const float* __restrict__ w1, const float* __restrict__ w2,
                        float* __restrict__ ca)
{
  __shared__ float hid[32];
  int b = blockIdx.x; int tid = threadIdx.x;
  if (tid < 16){
    float am=0.0f, ax=0.0f;
    for (int c=0;c<HID;c++){ float w=w1[tid*HID+c]; am+=mc[b*HID+c]*w; ax+=mx[b*HID+c]*w; }
    hid[tid]=fmaxf(am,0.0f); hid[16+tid]=fmaxf(ax,0.0f);
  }
  __syncthreads();
  float acc=0.0f;
  #pragma unroll
  for (int j=0;j<16;j++) acc += (hid[j]+hid[16+j]) * w2[tid*16+j];
  ca[b*HID+tid]=sigmoidf_(acc);
}

__global__ void k_spstat(const float* __restrict__ ln, const float* __restrict__ ca,
                         float* __restrict__ spm, float* __restrict__ spx)
{
  __shared__ float ssum[256]; __shared__ float smax[256];
  int bt = blockIdx.x;
  int b = bt / TB;
  int c = threadIdx.x;
  float v = ln[(size_t)bt*HID + c] * ca[b*HID+c];
  ssum[c]=v; smax[c]=v; __syncthreads();
  for (int st=128; st>0; st>>=1){
    if (c<st){ ssum[c]+=ssum[c+st]; smax[c]=fmaxf(smax[c],smax[c+st]); }
    __syncthreads();
  }
  if (c==0){ spm[bt]=ssum[0]*(1.0f/HID); spx[bt]=smax[0]; }
}

__global__ void k_cbfinal(const float* __restrict__ ln, const float* __restrict__ ca,
                          const float* __restrict__ spm, const float* __restrict__ spx,
                          const float* __restrict__ wsp, float* __restrict__ hbuf)
{
  __shared__ float sa_sh;
  int bt = blockIdx.x; int b = bt / TB; int t = bt % TB; int c = threadIdx.x;
  if (c==0){
    float acc=0.0f;
    #pragma unroll
    for (int j=0;j<7;j++){
      int tt=t-3+j;
      if(tt>=0&&tt<TB) acc += spm[b*TB+tt]*wsp[j] + spx[b*TB+tt]*wsp[7+j];
    }
    sa_sh = sigmoidf_(acc);
  }
  __syncthreads();
  size_t idx = (size_t)bt*HID + c;
  hbuf[idx] = ln[idx]*ca[b*HID+c]*sa_sh + hbuf[idx];
}

__global__ void k_rowmean(const float* __restrict__ h, float* __restrict__ hm)
{
  int b = blockIdx.x; int c = threadIdx.x;
  float s=0.0f;
  for (int t=0;t<TB;t++) s += h[((size_t)(b*TB+t))*HID + c];
  hm[b*HID+c] = s*(1.0f/TB);
}

__global__ void k_proj(const float* __restrict__ hm, const float* __restrict__ pw,
                       const float* __restrict__ pb, float* __restrict__ out)
{
  int idx = threadIdx.x;
  if (idx >= 40) return;
  int b = idx/10, j = idx%10;
  float acc = pb[j];
  for (int c=0;c<HID;c++) acc += hm[b*HID+c]*pw[j*HID+c];
  out[idx] = acc;
}

// ---------------------------------------------------------------------------
extern "C" void kernel_launch(void* const* d_in, const int* in_sizes, int n_in,
                              void* d_out, int out_size, void* d_ws, size_t ws_size,
                              hipStream_t stream)
{
  (void)in_sizes; (void)n_in; (void)out_size; (void)ws_size;
  const float* x     = (const float*)d_in[0];
  const float* w3    = (const float*)d_in[1];
  const float* b3    = (const float*)d_in[2];
  const float* w5    = (const float*)d_in[3];
  const float* b5    = (const float*)d_in[4];
  const float* w7    = (const float*)d_in[5];
  const float* b7    = (const float*)d_in[6];
  const float* in_w  = (const float*)d_in[7];
  const float* cmw   = (const float*)d_in[8];
  const float* cmb   = (const float*)d_in[9];
  const float* x_w   = (const float*)d_in[10];
  const float* dt_w  = (const float*)d_in[11];
  const float* dt_b  = (const float*)d_in[12];
  const float* A_log = (const float*)d_in[13];
  const float* Dp    = (const float*)d_in[14];
  const float* out_w = (const float*)d_in[15];
  const float* ng    = (const float*)d_in[16];
  const float* nb    = (const float*)d_in[17];
  const float* cw1   = (const float*)d_in[18];
  const float* cw2   = (const float*)d_in[19];
  const float* csp   = (const float*)d_in[20];
  const float* pw    = (const float*)d_in[21];
  const float* pb    = (const float*)d_in[22];
  float* out = (float*)d_out;

  float* ws = (float*)d_ws;
  size_t off = 0;
  float* buf_h    = ws + off; off += (size_t)ROWS*HID;      // f32
  float* buf_ln   = ws + off; off += (size_t)ROWS*HID;
  float* buf_tmp  = ws + off; off += (size_t)ROWS*HID;
  float* buf_xz   = ws + off; off += (size_t)ROWS*2*DIN;
  float* buf_xc   = ws + off; off += (size_t)ROWS*DIN;
  float* buf_dt   = ws + off; off += (size_t)ROWS*DIN;
  float* buf_yz   = ws + off; off += (size_t)ROWS*DIN;
  float* buf_xdbl = ws + off; off += (size_t)ROWS*48;
  float* buf_mc   = ws + off; off += BATCH*HID;
  float* buf_mx   = ws + off; off += BATCH*HID;
  float* buf_ca   = ws + off; off += BATCH*HID;
  float* buf_spm  = ws + off; off += BATCH*TB;
  float* buf_spx  = ws + off; off += BATCH*TB;
  float* buf_hm   = ws + off; off += BATCH*HID;
  // f16 staging buffers (carved from float workspace; all 16B aligned)
  _Float16* h16   = (_Float16*)(ws + off); off += (size_t)ROWS*HID/2;       // 1M halves
  _Float16* yz16  = (_Float16*)(ws + off); off += (size_t)ROWS*DIN/2;       // 2M halves
  _Float16* w16a  = (_Float16*)(ws + off); off += (size_t)(2*DIN)*HID/2;    // 256K halves
  _Float16* w16b  = (_Float16*)(ws + off); off += (size_t)HID*DIN/2;        // 128K halves

  dim3 blk(256);

  k_front<<<dim3((ROWS*HID+255)/256), blk, 0, stream>>>(x,w3,b3,w5,b5,w7,b7,buf_h);

  for (int i=0;i<5;i++){
    // --- f16 conversions for async WMMA GEMMs ---
    k_cvt16<<<dim3(ROWS*HID/8/256), blk, 0, stream>>>(buf_h, h16, ROWS*HID);
    k_cvt16<<<dim3(2*DIN*HID/8/256), blk, 0, stream>>>(in_w + (size_t)i*2*DIN*HID, w16a, 2*DIN*HID);
    // xz = h @ in_w^T   [4096,1024]  (async double-buffered WMMA)
    k_gemm_async<<<dim3(ROWS/64, (2*DIN)/128), blk, 0, stream>>>(
        h16, w16a, nullptr, buf_xz, ROWS, 2*DIN, HID);
    // depthwise conv + silu
    k_mconv<<<dim3(ROWS*DIN/256), blk, 0, stream>>>(
        buf_xz, cmw + (size_t)i*DIN*4, cmb + (size_t)i*DIN, buf_xc);
    // xdbl = xc @ x_w^T  [4096,48]  (ragged N -> sync WMMA with guards)
    k_gemm_sync<<<dim3(ROWS/64, 1), blk, 0, stream>>>(
        buf_xc, x_w + (size_t)i*48*DIN, buf_xdbl, ROWS, 48, DIN);
    // dt = softplus(...)
    k_dt<<<dim3(ROWS*DIN/256), blk, 0, stream>>>(
        buf_xdbl, dt_w + (size_t)i*DIN*16, dt_b + (size_t)i*DIN, buf_dt);
    // selective scan (fused with +xc*Dp and *silu(z))
    k_scan<<<dim3(BATCH*DIN/16), blk, 0, stream>>>(
        buf_xz, buf_xc, buf_dt, buf_xdbl,
        A_log + (size_t)i*DIN*16, Dp + (size_t)i*DIN, buf_yz);
    // tmp = h + yz @ out_w^T  (residual fused as WMMA C-input)
    k_cvt16<<<dim3(ROWS*DIN/8/256), blk, 0, stream>>>(buf_yz, yz16, ROWS*DIN);
    k_cvt16<<<dim3(HID*DIN/8/256), blk, 0, stream>>>(out_w + (size_t)i*HID*DIN, w16b, HID*DIN);
    k_gemm_async<<<dim3(ROWS/64, HID/128), blk, 0, stream>>>(
        yz16, w16b, buf_h, buf_tmp, ROWS, HID, DIN);
    // layernorm
    k_ln<<<dim3(ROWS), blk, 0, stream>>>(buf_tmp, ng + (size_t)i*HID, nb + (size_t)i*HID, buf_ln);
    // CBAM
    k_chstat<<<dim3(BATCH*HID), blk, 0, stream>>>(buf_ln, buf_mc, buf_mx);
    k_camlp<<<dim3(BATCH), blk, 0, stream>>>(buf_mc, buf_mx,
        cw1 + (size_t)i*16*HID, cw2 + (size_t)i*HID*16, buf_ca);
    k_spstat<<<dim3(ROWS), blk, 0, stream>>>(buf_ln, buf_ca, buf_spm, buf_spx);
    k_cbfinal<<<dim3(ROWS), blk, 0, stream>>>(buf_ln, buf_ca, buf_spm, buf_spx,
        csp + (size_t)i*14, buf_h);
  }

  k_rowmean<<<dim3(BATCH), blk, 0, stream>>>(buf_h, buf_hm);
  k_proj<<<dim3(1), dim3(64), 0, stream>>>(buf_hm, pw, pb, out);
}
